// Unified_RQVAE_76115410420025
// MI455X (gfx1250) — matrix operand order repaired and checked by reading, not compiled, not run on hardware
//
#include <hip/hip_runtime.h>
#include <math.h>

// ---- CDNA5 WMMA vector types -------------------------------------------------
typedef __attribute__((ext_vector_type(16))) __bf16 v16bf;
typedef __attribute__((ext_vector_type(8)))  __bf16 v8bf;
typedef __attribute__((ext_vector_type(8)))  float  v8f;

#define B_ROWS 4096
#define D_DIM  768
#define K_NEG  60
#define N_LOGITS (K_NEG + 1)     // 61 valid logits; 64 computed (3 padded)
#define KSTEPS   (D_DIM / 32)    // 24 K-steps of 32 dims

#if defined(__has_builtin) && __has_builtin(__builtin_amdgcn_sched_barrier)
#define SCHED_FENCE() __builtin_amdgcn_sched_barrier(0)
#else
#define SCHED_FENCE() do {} while (0)
#endif

// round-to-nearest-even f32 -> bf16 bit pattern
__device__ __forceinline__ unsigned short f32_to_bf16(float f) {
  unsigned u = __float_as_uint(f);
  u += 0x7FFFu + ((u >> 16) & 1u);
  return (unsigned short)(u >> 16);
}

// ---- Kernel A: convert the in-batch item table to bf16 (halves L2 traffic) --
__global__ void cvt_items_kernel(const float* __restrict__ src,
                                 unsigned short* __restrict__ dst) {
  int i = (blockIdx.x * blockDim.x + threadIdx.x) * 4;
  float4 v = *(const float4*)(src + i);
  ushort4 h;
  h.x = f32_to_bf16(v.x); h.y = f32_to_bf16(v.y);
  h.z = f32_to_bf16(v.z); h.w = f32_to_bf16(v.w);
  *(ushort4*)(dst + i) = h;
}

// ---- Kernel B: per-row gathered dots via v_wmma_f32_16x16x32_bf16 ------------
// One wave32 per row. 4 tiles of 16 items (slot 0 = positive, 1..60 = negatives,
// 61..63 = padding excluded from the softmax). Double-buffered K loop with
// sched_barrier fences so the backend cannot sink the prefetch loads back to
// their uses: stage kb+1's 8 global + 2 DS loads stay in flight while the
// matrix pipe consumes stage kb.
__global__ void __launch_bounds__(256)
u2i_loss_kernel(const float* __restrict__ user_emb,
                const unsigned short* __restrict__ items_bf,
                const int* __restrict__ neg_idx,
                float* __restrict__ losses) {
  __shared__ __align__(16) unsigned short uL[8][D_DIM]; // per-wave user row, bf16
  __shared__ float lg[8][64];                           // per-wave logits

  const int lane = threadIdx.x & 31;
  const int w    = threadIdx.x >> 5;
  const int b    = blockIdx.x * 8 + w;

  // 1) convert this row's user embedding to bf16 in LDS (B operand source)
  const float* urow = user_emb + (size_t)b * D_DIM;
  #pragma unroll
  for (int c = 0; c < 6; ++c) {
    int e = c * 128 + lane * 4;
    float4 v = *(const float4*)(urow + e);
    ushort4 h;
    h.x = f32_to_bf16(v.x); h.y = f32_to_bf16(v.y);
    h.z = f32_to_bf16(v.z); h.w = f32_to_bf16(v.w);
    *(ushort4*)&uL[w][e] = h;
  }
  __syncthreads();

  // 2) gathered item base pointers, one item per lane-pair (A-matrix rows)
  const int j    = lane & 15;   // item within tile (M index)
  const int half = lane >> 4;   // 16-bit A layout: half 0 -> K0-7/16-23, half 1 -> K8-15/24-31
  const int nb   = b * K_NEG;
  int idx[4];
  idx[0] = (j == 0)  ? b : neg_idx[nb + j - 1];
  idx[1] = neg_idx[nb + 15 + j];
  idx[2] = neg_idx[nb + 31 + j];
  idx[3] = (j <= 12) ? neg_idx[nb + 47 + j] : b;  // slots 61..63: pad (masked later)

  const unsigned short* abase[4];
  #pragma unroll
  for (int t = 0; t < 4; ++t)
    abase[t] = items_bf + (size_t)idx[t] * D_DIM + half * 8;

  const unsigned short* bbase = &uL[w][half * 8];

  // 3) K loop: 24 steps of K=32, software-pipelined with 2 register buffers.
  v8bf Alo[2][4], Ahi[2][4], Blo[2], Bhi[2];
  v8f  acc[4] = {v8f{}, v8f{}, v8f{}, v8f{}};

#define LOAD_STEP(buf, kb_)                                   \
  do {                                                        \
    const int ke_ = (kb_) * 32;                               \
    Blo[buf] = *(const v8bf*)(bbase + ke_);                   \
    Bhi[buf] = *(const v8bf*)(bbase + ke_ + 16);              \
    _Pragma("unroll")                                         \
    for (int t = 0; t < 4; ++t) {                             \
      Alo[buf][t] = *(const v8bf*)(abase[t] + ke_);           \
      Ahi[buf][t] = *(const v8bf*)(abase[t] + ke_ + 16);      \
    }                                                         \
  } while (0)

  LOAD_STEP(0, 0);
  SCHED_FENCE();
  #pragma unroll
  for (int kb = 0; kb < KSTEPS; ++kb) {
    const int cur = kb & 1;
    const int nxt = cur ^ 1;
    if (kb + 1 < KSTEPS) LOAD_STEP(nxt, kb + 1);   // prefetch next step
    SCHED_FENCE();  // keep prefetch ABOVE the consumes of stage `cur`
    v16bf Bm = __builtin_shufflevector(Blo[cur], Bhi[cur],
        0,1,2,3,4,5,6,7,8,9,10,11,12,13,14,15);
    #pragma unroll
    for (int t = 0; t < 4; ++t) {
      v16bf Am = __builtin_shufflevector(Alo[cur][t], Ahi[cur][t],
          0,1,2,3,4,5,6,7,8,9,10,11,12,13,14,15);
      acc[t] = __builtin_amdgcn_wmma_f32_16x16x32_bf16(
          false, Am, false, Bm, (short)0, acc[t], false, false);
    }
    SCHED_FENCE();  // keep stage-cur WMMAs ahead of next iteration's prefetch
  }
#undef LOAD_STEP

  // 4) all N columns identical; row m sits in VGPR m%8, lane-half m/8.
  if (j == 0) {  // lanes 0 and 16
    #pragma unroll
    for (int t = 0; t < 4; ++t) {
      #pragma unroll
      for (int i = 0; i < 8; ++i)
        lg[w][t * 16 + half * 8 + i] = acc[t][i];
    }
  }
  __syncthreads();

  // 5) softmax CE over the 61 valid logits (label 0), wave32 shuffle reduction
  float x0 = lg[w][lane];
  float x1 = (lane + 32 < N_LOGITS) ? lg[w][lane + 32] : -INFINITY;
  float m = fmaxf(x0, x1);
  #pragma unroll
  for (int off = 16; off; off >>= 1)
    m = fmaxf(m, __shfl_xor(m, off, 32));
  float e = __expf(x0 - m) + ((lane + 32 < N_LOGITS) ? __expf(x1 - m) : 0.0f);
  #pragma unroll
  for (int off = 16; off; off >>= 1)
    e += __shfl_xor(e, off, 32);
  if (lane == 0)
    losses[b] = m + __logf(e) - lg[w][0];
}

// ---- Kernel C: deterministic tree reduction -> mean --------------------------
__global__ void reduce_mean_kernel(const float* __restrict__ losses,
                                   float* __restrict__ out) {
  __shared__ float s[256];
  int tid = threadIdx.x;
  float a = 0.0f;
  for (int i = tid; i < B_ROWS; i += 256) a += losses[i];
  s[tid] = a;
  __syncthreads();
  for (int off = 128; off; off >>= 1) {
    if (tid < off) s[tid] += s[tid + off];
    __syncthreads();
  }
  if (tid == 0) out[0] = s[0] * (1.0f / (float)B_ROWS);
}

// ---- Host launch -------------------------------------------------------------
extern "C" void kernel_launch(void* const* d_in, const int* in_sizes, int n_in,
                              void* d_out, int out_size, void* d_ws, size_t ws_size,
                              hipStream_t stream) {
  const float* user_emb = (const float*)d_in[0];
  const float* item_emb = (const float*)d_in[1];
  const int*   neg_idx  = (const int*)d_in[2];

  unsigned short* items_bf = (unsigned short*)d_ws;                      // 6.29 MB
  float* losses = (float*)((char*)d_ws +
                  (size_t)B_ROWS * D_DIM * sizeof(unsigned short));      // 16 KB

  // A: f32 -> bf16 item table (3.1M elems, 4 per thread)
  cvt_items_kernel<<<(B_ROWS * D_DIM) / 1024, 256, 0, stream>>>(item_emb, items_bf);
  // B: one wave per row, WMMA gathered dots + per-row loss
  u2i_loss_kernel<<<B_ROWS / 8, 256, 0, stream>>>(user_emb, items_bf, neg_idx, losses);
  // C: mean
  reduce_mean_kernel<<<1, 256, 0, stream>>>(losses, (float*)d_out);
}
